// AttentionPITF_41790031790227
// MI455X (gfx1250) — compile-verified, matrix operand order checked by reading, not run on hardware
//
#include <hip/hip_runtime.h>

// ---------------- WMMA types ----------------
typedef __attribute__((ext_vector_type(16))) __bf16 v16bf;
typedef __attribute__((ext_vector_type(8)))  float  v8f;

union Frag {
    unsigned int u[8];
    uint4 q[2];
    v16bf v;
};

static __device__ __forceinline__ v8f wmma_bf16(v16bf a, v16bf b, v8f c) {
    return __builtin_amdgcn_wmma_f32_16x16x32_bf16(false, a, false, b, (short)0, c, false, false);
}

// B-tile (32x16) lane fragment from a pre-converted bf16 weight row.
// kbase = kb*32 + (lane<16 ? 0 : 8); needs K {kbase..kbase+7, kbase+16..kbase+23}.
static __device__ __forceinline__ v16bf load_b_bf16(const __bf16* rowPtr, int kbase) {
    const uint4* p = reinterpret_cast<const uint4*>(rowPtr + kbase);
    Frag f;
    f.q[0] = p[0];   // K kbase..kbase+7
    f.q[1] = p[2];   // K kbase+16..kbase+23
    return f.v;
}

// Fallback: build the same fragment from f32 weights (native cvt instructions).
static __device__ __forceinline__ v16bf load_b_f32(const float* rowPtr, int kbase) {
    const float4* p = reinterpret_cast<const float4*>(rowPtr + kbase);
    float4 a0 = p[0], a1 = p[1], a2 = p[4], a3 = p[5];
    v16bf r;
    r[0] = (__bf16)a0.x; r[1] = (__bf16)a0.y; r[2]  = (__bf16)a0.z; r[3]  = (__bf16)a0.w;
    r[4] = (__bf16)a1.x; r[5] = (__bf16)a1.y; r[6]  = (__bf16)a1.z; r[7]  = (__bf16)a1.w;
    r[8] = (__bf16)a2.x; r[9] = (__bf16)a2.y; r[10] = (__bf16)a2.z; r[11] = (__bf16)a2.w;
    r[12] = (__bf16)a3.x; r[13] = (__bf16)a3.y; r[14] = (__bf16)a3.z; r[15] = (__bf16)a3.w;
    return r;
}

// A-tile (16x32) lane fragment from LDS, row-major bf16 packed in uints.
static __device__ __forceinline__ v16bf load_a_lds(const unsigned int* base, int rowStrideU,
                                                   int row, int kb, int lane) {
    int idx = row * rowStrideU + kb * 16 + ((lane < 16) ? 0 : 4);
    Frag f;
    f.q[0] = *reinterpret_cast<const uint4*>(&base[idx]);
    f.q[1] = *reinterpret_cast<const uint4*>(&base[idx + 8]);
    return f.v;
}

// ---------------- Problem constants ----------------
#define BATCH      16384
#define KDIM       256
#define MHIST      50
#define MPAD       64
#define XCOLS      54         // 4 + M
#define CATK       1024
#define EPB        16         // batch rows per block
#define TPB        128        // 4 waves

#define HROWU      132        // uints per padded hist row (264 bf16, 528B)
#define HROWH      264
#define CROWU      516        // uints per padded cat row (1032 bf16)
#define DUTROW     260        // floats per padded (ut-nut) row

#define WATT_N     (KDIM * KDIM)        // 65536
#define WMAP_N     (KDIM * CATK)        // 262144
#define WS_NEEDED  ((size_t)(WATT_N + WMAP_N) * 2)

#define SMEM_BYTES (64*HROWU*4 + EPB*CROWU*4 + EPB*DUTROW*4 + 256*4 + 64*4 + 64*4 + 16*4 + 64*4)

// ---------------- Weight pre-conversion (f32 -> bf16 in workspace) ----------------
__global__ void convert_weights_kernel(const float* __restrict__ Watt,
                                       const float* __restrict__ Wmap,
                                       __bf16* __restrict__ dst) {
    int i = blockIdx.x * blockDim.x + threadIdx.x;
    int stride = gridDim.x * blockDim.x;
    for (; i < WATT_N + WMAP_N; i += stride) {
        float v = (i < WATT_N) ? Watt[i] : Wmap[i - WATT_N];
        dst[i] = (__bf16)v;
    }
}

// ---------------- Fused attention + mapping kernel ----------------
template <bool PRECONV>
__global__ __launch_bounds__(TPB) void attn_pitf_kernel(
    const int* __restrict__ x,
    const float* __restrict__ userV,
    const float* __restrict__ itemV,
    const float* __restrict__ tagU,
    const float* __restrict__ tagI,
    const float* __restrict__ Watt,
    const float* __restrict__ batt,
    const float* __restrict__ Wmap,
    const float* __restrict__ bmap,
    const __bf16* __restrict__ wbf,     // [WATT_N] ++ [WMAP_N] when PRECONV
    float* __restrict__ out)
{
    extern __shared__ unsigned char smem[];
    unsigned int* sHist  = reinterpret_cast<unsigned int*>(smem);        // 64 x HROWU
    unsigned int* sCat   = sHist + 64 * HROWU;                           // 16 x CROWU
    float* sDut   = reinterpret_cast<float*>(sCat + EPB * CROWU);        // 16 x DUTROW
    float* sU     = sDut + EPB * DUTROW;                                 // 256
    float* sScore = sU + 256;                                            // 64
    float* sAlpha = sScore + 64;                                         // 64
    float* sR     = sAlpha + 64;                                         // 16
    int*   sHidx  = reinterpret_cast<int*>(sR + 16);                     // 64
    __bf16* sHistB = reinterpret_cast<__bf16*>(sHist);

    const int t    = threadIdx.x;
    const int lane = t & 31;
    const int wave = t >> 5;
    const int b0   = blockIdx.x * EPB;

    const __bf16* WattB = wbf;
    const __bf16* WmapB = wbf + WATT_N;

    if (t < EPB) sR[t] = 0.0f;
    __syncthreads();

    // =============== Per-batch-row phase: attention ===============
    for (int e = 0; e < EPB; ++e) {
        const int b = b0 + e;
        const int* xb = x + (size_t)b * XCOLS;
        const int i0 = xb[0], i1 = xb[1], i2 = xb[2], i3 = xb[3];

        if (t < MHIST) sHidx[t] = xb[4 + t];
        if (t < MPAD)  sScore[t] = 0.0f;
        __syncthreads();

        // Gather hist rows -> LDS bf16 (pad rows 50..63 with zeros)
        for (int idx = t; idx < MPAD * KDIM; idx += TPB) {
            int m = idx >> 8, k = idx & 255;
            float v = (m < MHIST) ? tagU[(size_t)sHidx[m] * KDIM + k] : 0.0f;
            sHistB[m * HROWH + k] = (__bf16)v;
        }
        // u, (ut-nut), and iv*(it-nit) partial
        float ivp = 0.0f;
        {
            const float* uvr  = userV + (size_t)i0 * KDIM;
            const float* utr  = tagU  + (size_t)i2 * KDIM;
            const float* nutr = tagU  + (size_t)i3 * KDIM;
            const float* ivr  = itemV + (size_t)i1 * KDIM;
            const float* itr  = tagI  + (size_t)i2 * KDIM;
            const float* nitr = tagI  + (size_t)i3 * KDIM;
            #pragma unroll
            for (int h = 0; h < 2; ++h) {
                int k = t + h * TPB;
                sU[k] = uvr[k];
                sDut[e * DUTROW + k] = utr[k] - nutr[k];
                ivp += ivr[k] * (itr[k] - nitr[k]);
            }
        }
        {   // wave reduce + accumulate iv term into sR[e]
            float v = ivp;
            #pragma unroll
            for (int msk = 1; msk < 32; msk <<= 1) v += __shfl_xor(v, msk);
            if (lane == 0) atomicAdd(&sR[e], v);
        }
        __syncthreads();

        // --- GEMM1: scores[m] = sum_j relu(hist@Watt^T + b_att)[m,j] * u[j]
        float sp[4][8];
        #pragma unroll
        for (int mt = 0; mt < 4; ++mt)
            #pragma unroll
            for (int i = 0; i < 8; ++i) sp[mt][i] = 0.0f;

        for (int jj = 0; jj < 4; ++jj) {
            const int jt  = wave * 4 + jj;
            const int col = jt * 16 + (lane & 15);
            const float uj = sU[col];
            const float ba = batt[col];
            const int kshift = (lane < 16) ? 0 : 8;
            const __bf16* wrowB = WattB + (size_t)col * KDIM;
            const float*  wrowF = Watt  + (size_t)col * KDIM;

            v8f acc[4];
            #pragma unroll
            for (int mt = 0; mt < 4; ++mt) acc[mt] = (v8f)0.0f;

            #pragma unroll 2
            for (int kb = 0; kb < 8; ++kb) {
                v16bf bt = PRECONV ? load_b_bf16(wrowB, kb * 32 + kshift)
                                   : load_b_f32 (wrowF, kb * 32 + kshift);
                #pragma unroll
                for (int mt = 0; mt < 4; ++mt) {
                    v16bf at = load_a_lds(sHist, HROWU, mt * 16 + (lane & 15), kb, lane);
                    acc[mt] = wmma_bf16(at, bt, acc[mt]);
                }
            }
            #pragma unroll
            for (int mt = 0; mt < 4; ++mt)
                #pragma unroll
                for (int i = 0; i < 8; ++i) {
                    float c = fmaxf(acc[mt][i] + ba, 0.0f);
                    sp[mt][i] += c * uj;
                }
        }
        // reduce over 16-lane column halves, accumulate into sScore
        #pragma unroll
        for (int mt = 0; mt < 4; ++mt)
            #pragma unroll
            for (int i = 0; i < 8; ++i) {
                float v = sp[mt][i];
                v += __shfl_xor(v, 1); v += __shfl_xor(v, 2);
                v += __shfl_xor(v, 4); v += __shfl_xor(v, 8);
                if ((lane & 15) == 0)
                    atomicAdd(&sScore[mt * 16 + ((lane >> 4) << 3) + i], v);
            }
        __syncthreads();

        // --- softmax over m<50 (wave 0)
        if (t < 32) {
            float s0 = sScore[t];
            int m1 = t + 32;
            float s1 = (m1 < MHIST) ? sScore[m1] : -3.0e38f;
            float mx = fmaxf(s0, s1);
            #pragma unroll
            for (int msk = 1; msk < 32; msk <<= 1) mx = fmaxf(mx, __shfl_xor(mx, msk));
            float e0 = __expf(s0 - mx);
            float e1 = (m1 < MHIST) ? __expf(s1 - mx) : 0.0f;
            float sm = e0 + e1;
            #pragma unroll
            for (int msk = 1; msk < 32; msk <<= 1) sm += __shfl_xor(sm, msk);
            float inv = 1.0f / sm;
            sAlpha[t]  = e0 * inv;
            sAlpha[m1] = e1 * inv;
        }
        __syncthreads();

        // --- h = sum_m alpha[m]*hist[m,:]  (f32 from global for precision), build cat row
        {
            float hv0 = 0.0f, hv1 = 0.0f;
            for (int m = 0; m < MHIST; ++m) {
                float a = sAlpha[m];
                const float* hr = tagU + (size_t)sHidx[m] * KDIM;
                hv0 += a * hr[t];
                hv1 += a * hr[t + TPB];
            }
            __bf16* catRow = reinterpret_cast<__bf16*>(sCat + e * CROWU);
            float u0 = sU[t], u1 = sU[t + TPB];
            catRow[t]             = (__bf16)u0;
            catRow[t + TPB]       = (__bf16)u1;
            catRow[256 + t]       = (__bf16)hv0;
            catRow[256 + t + TPB] = (__bf16)hv1;
            catRow[512 + t]       = (__bf16)(u0 - hv0);
            catRow[512 + t + TPB] = (__bf16)(u1 - hv1);
            catRow[768 + t]       = (__bf16)(u0 * hv0);
            catRow[768 + t + TPB] = (__bf16)(u1 * hv1);
        }
        __syncthreads();
    }

    // =============== GEMM2: mix = relu(cat @ Wmap^T + b_map), fused r reduction
    {
        float rp[8];
        #pragma unroll
        for (int i = 0; i < 8; ++i) rp[i] = 0.0f;
        const int kshift = (lane < 16) ? 0 : 8;

        for (int jj = 0; jj < 4; ++jj) {
            const int jt  = wave * 4 + jj;
            const int col = jt * 16 + (lane & 15);
            const __bf16* wrowB = WmapB + (size_t)col * CATK;
            const float*  wrowF = Wmap  + (size_t)col * CATK;
            v8f acc = (v8f)0.0f;
            #pragma unroll 2
            for (int kb = 0; kb < 32; ++kb) {
                v16bf bt = PRECONV ? load_b_bf16(wrowB, kb * 32 + kshift)
                                   : load_b_f32 (wrowF, kb * 32 + kshift);
                v16bf at = load_a_lds(sCat, CROWU, (lane & 15), kb, lane);
                acc = wmma_bf16(at, bt, acc);
            }
            const float bm = bmap[col];
            #pragma unroll
            for (int i = 0; i < 8; ++i) {
                float c = fmaxf(acc[i] + bm, 0.0f);
                int row = ((lane < 16) ? 0 : 8) + i;   // batch-local row
                rp[i] += c * sDut[row * DUTROW + col];
            }
        }
        #pragma unroll
        for (int i = 0; i < 8; ++i) {
            float v = rp[i];
            v += __shfl_xor(v, 1); v += __shfl_xor(v, 2);
            v += __shfl_xor(v, 4); v += __shfl_xor(v, 8);
            if ((lane & 15) == 0)
                atomicAdd(&sR[((lane >> 4) << 3) + i], v);
        }
    }
    __syncthreads();
    if (t < EPB) out[b0 + t] = sR[t];
}

extern "C" void kernel_launch(void* const* d_in, const int* in_sizes, int n_in,
                              void* d_out, int out_size, void* d_ws, size_t ws_size,
                              hipStream_t stream) {
    (void)in_sizes; (void)n_in; (void)out_size;
    const int*   x     = (const int*)  d_in[0];
    const float* userV = (const float*)d_in[1];
    const float* itemV = (const float*)d_in[2];
    const float* tagU  = (const float*)d_in[3];
    const float* tagI  = (const float*)d_in[4];
    const float* Watt  = (const float*)d_in[5];
    const float* batt  = (const float*)d_in[6];
    const float* Wmap  = (const float*)d_in[7];
    const float* bmap  = (const float*)d_in[8];
    float* out = (float*)d_out;

    dim3 grid(BATCH / EPB), block(TPB);
    if (ws_size >= WS_NEEDED && d_ws != nullptr) {
        __bf16* wbf = (__bf16*)d_ws;
        convert_weights_kernel<<<dim3(512), dim3(256), 0, stream>>>(Watt, Wmap, wbf);
        attn_pitf_kernel<true><<<grid, block, SMEM_BYTES, stream>>>(
            x, userV, itemV, tagU, tagI, Watt, batt, Wmap, bmap, wbf, out);
    } else {
        attn_pitf_kernel<false><<<grid, block, SMEM_BYTES, stream>>>(
            x, userV, itemV, tagU, tagI, Watt, batt, Wmap, bmap, nullptr, out);
    }
}
